// MutiHeadAttentionV1_9311489097997
// MI455X (gfx1250) — matrix operand order repaired and checked by reading, not compiled, not run on hardware
//
#include <hip/hip_runtime.h>
#include <hip/hip_bf16.h>

// Problem constants (from reference)
#define BATCH    2
#define SEQ      2048
#define DMODEL   1024
#define NHEAD    16
#define HDIM     64
#define MTOT     (BATCH * SEQ)   // 4096

typedef __attribute__((ext_vector_type(16))) __bf16 v16bf;
typedef __attribute__((ext_vector_type(4)))  __bf16 v4bf;
typedef __attribute__((ext_vector_type(8)))  float  v8f;

union FragBF { uint4 u[2]; v16bf v; };

static __device__ __forceinline__ v8f wmma_bf16(v16bf a, v16bf b, v8f c) {
    // D = A(16x32 bf16) * B(32x16 bf16) + C(16x16 f32)
    return __builtin_amdgcn_wmma_f32_16x16x32_bf16(
        false, a, false, b, (short)0, c, false, false);
}

static __device__ __forceinline__ v8f zero_v8f() {
    v8f z = {0.f, 0.f, 0.f, 0.f, 0.f, 0.f, 0.f, 0.f};
    return z;
}

// ---------------------------------------------------------------------------
// Async global->LDS copy of 16 bytes (one b128), with sync fallback.
// Builtin signature (from hipcc diagnostic): param0 is
//   '__attribute__((__vector_size__(4*sizeof(int)))) int __device__ *'
// i.e. int4* in addrspace(1); LDS side is int4* in addrspace(3).
// Flat LDS pointer bits [31:0] are the LDS byte offset (ISA 10.2 aperture),
// so we materialize the addrspace(3) pointer by 32-bit truncation.
// ---------------------------------------------------------------------------
#if defined(__has_builtin)
#if __has_builtin(__builtin_amdgcn_global_load_async_to_lds_b128)
#define HAVE_ASYNC_LDS 1
#endif
#endif

typedef int v4i_vec __attribute__((vector_size(16)));
typedef __attribute__((address_space(1))) v4i_vec* as1_v4i;
typedef __attribute__((address_space(3))) v4i_vec* as3_v4i;

static __device__ __forceinline__ void cp16(const __bf16* g, __bf16* l) {
#ifdef HAVE_ASYNC_LDS
    __builtin_amdgcn_global_load_async_to_lds_b128(
        (as1_v4i)(uintptr_t)g,
        (as3_v4i)(unsigned int)(uintptr_t)l,
        0, 0);
#else
    *reinterpret_cast<uint4*>(l) = *reinterpret_cast<const uint4*>(g);
#endif
}

static __device__ __forceinline__ void wait_async_lds() {
#ifdef HAVE_ASYNC_LDS
#if __has_builtin(__builtin_amdgcn_s_wait_asynccnt)
    __builtin_amdgcn_s_wait_asynccnt(0);
#else
    asm volatile("s_wait_asynccnt 0" ::: "memory");
#endif
#endif
}

// ---------------------------------------------------------------------------
// Kernel 1: cast f32 activations -> bf16, 4 elements/thread
// ---------------------------------------------------------------------------
__global__ __launch_bounds__(256)
void cast_in_kernel(const float* __restrict__ in, __bf16* __restrict__ out, int n4) {
    int i = blockIdx.x * blockDim.x + threadIdx.x;
    if (i >= n4) return;
    float4 f = reinterpret_cast<const float4*>(in)[i];
    v4bf h;
    h[0] = (__bf16)f.x; h[1] = (__bf16)f.y; h[2] = (__bf16)f.z; h[3] = (__bf16)f.w;
    reinterpret_cast<v4bf*>(out)[i] = h;
}

// ---------------------------------------------------------------------------
// Kernel 2: transpose-cast weights W[K][N] f32 -> Wt[N][K] bf16 (z picks matrix)
// ---------------------------------------------------------------------------
__global__ __launch_bounds__(256)
void wtrans_kernel(const float* __restrict__ w0, const float* __restrict__ w1,
                   const float* __restrict__ w2, const float* __restrict__ w3,
                   __bf16* __restrict__ t0, __bf16* __restrict__ t1,
                   __bf16* __restrict__ t2, __bf16* __restrict__ t3) {
    const float* src; __bf16* dst;
    switch (blockIdx.z) {
        case 0:  src = w0; dst = t0; break;
        case 1:  src = w1; dst = t1; break;
        case 2:  src = w2; dst = t2; break;
        default: src = w3; dst = t3; break;
    }
    int k = blockIdx.x * blockDim.x + threadIdx.x;   // fast dim -> coalesced writes
    int n = blockIdx.y;
    dst[n * DMODEL + k] = (__bf16)src[k * DMODEL + n];
}

// ---------------------------------------------------------------------------
// Shared GEMM tiling constants: block computes 128(M) x 128(N), k-step = 32.
// LDS tiles padded to stride 40 elems (80B) -> conflict-free ds_load_b128.
// ---------------------------------------------------------------------------
#define KSTEP   32
#define LDSTR   40                    // padded row stride in elements
#define TILE_E  (128 * LDSTR)         // elements per LDS tile buffer

// ---------------------------------------------------------------------------
// Kernel 3: fused QKV projection, LDS double-buffered.
//   X[4096,1024] bf16 @ Wt[1024(n),1024(k)] -> Q/K: [B,H,S,Dh], V: [B,H,Dh,S]
// ---------------------------------------------------------------------------
__global__ __launch_bounds__(256)
void qkv_gemm_kernel(const __bf16* __restrict__ X,
                     const __bf16* __restrict__ Wqt, const __bf16* __restrict__ Wkt,
                     const __bf16* __restrict__ Wvt,
                     __bf16* __restrict__ Q, __bf16* __restrict__ Km,
                     __bf16* __restrict__ Vt) {
    __shared__ __align__(16) __bf16 As[2][TILE_E];
    __shared__ __align__(16) __bf16 Bs[2][TILE_E];

    const int tid  = threadIdx.x;
    const int wid  = tid >> 5;
    const int lane = tid & 31;
    const int half = lane >> 4;
    const int lm   = lane & 15;
    const int wm   = wid >> 1;        // 0..3 -> 2 m-tiles each
    const int wn   = wid & 1;         // 0..1 -> 4 n-tiles each

    const int which = blockIdx.z;     // 0=Q, 1=K, 2=V
    const __bf16* Wt = (which == 0) ? Wqt : ((which == 1) ? Wkt : Wvt);

    const int m0 = blockIdx.x * 128;
    const int n0 = blockIdx.y * 128;
    const __bf16* Ag = X  + (size_t)m0 * DMODEL;
    const __bf16* Bg = Wt + (size_t)n0 * DMODEL;

    // stage helper: 256 threads x (2 A-chunks + 2 B-chunks) of 16B
    auto stage = [&](int buf, int k0) {
        for (int i = 0; i < 2; ++i) {
            int t   = tid * 2 + i;            // 0..511
            int row = t >> 2, c = t & 3;      // 128 rows x 4 chunks
            cp16(Ag + (size_t)row * DMODEL + k0 + c * 8, &As[buf][row * LDSTR + c * 8]);
            cp16(Bg + (size_t)row * DMODEL + k0 + c * 8, &Bs[buf][row * LDSTR + c * 8]);
        }
    };

    v8f acc[2][4];
    for (int i = 0; i < 2; ++i)
        for (int j = 0; j < 4; ++j) acc[i][j] = zero_v8f();

    stage(0, 0);
    for (int ks = 0; ks < DMODEL / KSTEP; ++ks) {
        int cur = ks & 1;
        wait_async_lds();
        __syncthreads();                       // buf 'cur' visible to all waves
        if (ks + 1 < DMODEL / KSTEP) stage(1 - cur, (ks + 1) * KSTEP);

        const __bf16* ab = &As[cur][0];
        const __bf16* bb = &Bs[cur][0];
        FragBF a[2], b[4];
        for (int i = 0; i < 2; ++i) {
            int row = ((wm * 2 + i) * 16 + lm) * LDSTR;
            a[i].u[0] = *reinterpret_cast<const uint4*>(ab + row + half * 16);
            a[i].u[1] = *reinterpret_cast<const uint4*>(ab + row + half * 16 + 8);
        }
        for (int j = 0; j < 4; ++j) {
            int row = ((wn * 4 + j) * 16 + lm) * LDSTR;
            b[j].u[0] = *reinterpret_cast<const uint4*>(bb + row + half * 16);
            b[j].u[1] = *reinterpret_cast<const uint4*>(bb + row + half * 16 + 8);
        }
        for (int i = 0; i < 2; ++i)
            for (int j = 0; j < 4; ++j)
                acc[i][j] = wmma_bf16(a[i].v, b[j].v, acc[i][j]);
        __syncthreads();                       // done reading buf 'cur'
    }

    const float scale = (which == 0) ? 0.125f : 1.0f;   // 1/sqrt(HDIM) into Q
    for (int i = 0; i < 2; ++i) {
        for (int j = 0; j < 4; ++j) {
            int n = n0 + (wn * 4 + j) * 16 + lm;
            int h = n >> 6, dh = n & 63;
            for (int r = 0; r < 8; ++r) {
                int m = m0 + (wm * 2 + i) * 16 + r + half * 8;
                int b_ = m >> 11, s = m & (SEQ - 1);
                float val = acc[i][j][r] * scale;
                if (which == 2) {
                    Vt[((size_t)(b_ * NHEAD + h) * HDIM + dh) * SEQ + s] = (__bf16)val;
                } else {
                    __bf16* dst = (which == 0) ? Q : Km;
                    dst[((size_t)(b_ * NHEAD + h) * SEQ + s) * HDIM + dh] = (__bf16)val;
                }
            }
        }
    }
}

// ---------------------------------------------------------------------------
// Kernel 4: causal flash attention. One wave per (b,h,qtile of 16 rows);
// all 8 waves of a block share one (b,h) so K/V tiles hit the WGP cache.
// ---------------------------------------------------------------------------
__global__ __launch_bounds__(256)
void attn_kernel(const __bf16* __restrict__ Q, const __bf16* __restrict__ Km,
                 const __bf16* __restrict__ Vt, __bf16* __restrict__ ctx) {
    __shared__ __align__(16) __bf16 Plds[8][16][32];   // per-wave P staging (8 KB)

    const int wid  = threadIdx.x >> 5;
    const int lane = threadIdx.x & 31;
    const int half = lane >> 4;
    const int lm   = lane & 15;

    int bh    = blockIdx.x >> 4;               // 0..31 == b*NHEAD + h
    int qtile = (blockIdx.x & 15) * 8 + wid;   // 0..127

    // Q A-fragments for dh 0..31 and 32..63 (loop invariant)
    const __bf16* qrow = Q + ((size_t)bh * SEQ + qtile * 16 + lm) * HDIM;
    FragBF qa[2];
    for (int s = 0; s < 2; ++s) {
        qa[s].u[0] = *reinterpret_cast<const uint4*>(qrow + s * 32 + half * 16);
        qa[s].u[1] = *reinterpret_cast<const uint4*>(qrow + s * 32 + half * 16 + 8);
    }

    float m_run[8], l_run[8];
    for (int r = 0; r < 8; ++r) { m_run[r] = -__builtin_inff(); l_run[r] = 0.f; }
    v8f o[4];
    for (int t = 0; t < 4; ++t) o[t] = zero_v8f();

    const int nIter = ((qtile + 1) * 16 + 31) >> 5;     // 32-key steps to diagonal
    for (int it = 0; it < nIter; ++it) {
        const int key0 = it * 32;

        // scores: two 16x16 tiles (keys key0..+15, key0+16..+31), k over dh
        v8f sc[2];
        sc[0] = zero_v8f(); sc[1] = zero_v8f();
        for (int c = 0; c < 2; ++c) {
            for (int s = 0; s < 2; ++s) {
                const __bf16* krow = Km + ((size_t)bh * SEQ + key0 + c * 16 + lm) * HDIM
                                        + s * 32 + half * 16;
                FragBF kb;   // B[dh][key] = K[key][dh]: contiguous per lane
                kb.u[0] = reinterpret_cast<const uint4*>(krow)[0];
                kb.u[1] = reinterpret_cast<const uint4*>(krow)[1];
                sc[c] = wmma_bf16(qa[s].v, kb.v, sc[c]);
            }
        }

        // V B-fragments issued EARLY so their latency overlaps the softmax math
        // (they must precede the asm dscnt fence, which has a memory clobber).
        FragBF vb[4];
        for (int t = 0; t < 4; ++t) {
            const __bf16* vrow = Vt + ((size_t)bh * HDIM + t * 16 + lm) * SEQ
                                    + key0 + half * 16;
            vb[t].u[0] = reinterpret_cast<const uint4*>(vrow)[0];
            vb[t].u[1] = reinterpret_cast<const uint4*>(vrow)[1];
        }

        // causal mask (Q already carries 1/sqrt(d) scale)
        for (int c = 0; c < 2; ++c) {
            int ncol = key0 + c * 16 + lm;
            for (int r = 0; r < 8; ++r) {
                int mrow = qtile * 16 + r + half * 8;
                if (ncol > mrow) sc[c][r] = -__builtin_inff();
            }
        }

        // online softmax (row stats replicated across each 16-lane group)
        float pr0[8], pr1[8];
        for (int r = 0; r < 8; ++r) {
            float t = fmaxf(sc[0][r], sc[1][r]);
            t = fmaxf(t, __shfl_xor(t, 1, 32));
            t = fmaxf(t, __shfl_xor(t, 2, 32));
            t = fmaxf(t, __shfl_xor(t, 4, 32));
            t = fmaxf(t, __shfl_xor(t, 8, 32));
            float mn = fmaxf(m_run[r], t);
            float f  = __expf(m_run[r] - mn);
            float p0 = __expf(sc[0][r] - mn);
            float p1 = __expf(sc[1][r] - mn);
            float su = p0 + p1;
            su += __shfl_xor(su, 1, 32);
            su += __shfl_xor(su, 2, 32);
            su += __shfl_xor(su, 4, 32);
            su += __shfl_xor(su, 8, 32);
            l_run[r] = l_run[r] * f + su;
            m_run[r] = mn;
            pr0[r] = p0; pr1[r] = p1;
            for (int t2 = 0; t2 < 4; ++t2) o[t2][r] *= f;
        }

        // transpose P (D-layout) -> A-layout through LDS
        for (int r = 0; r < 8; ++r) {
            Plds[wid][r + half * 8][lm]      = (__bf16)pr0[r];
            Plds[wid][r + half * 8][16 + lm] = (__bf16)pr1[r];
        }
        asm volatile("s_wait_dscnt 0" ::: "memory");   // wave-local LDS RAW fence
        FragBF pa;   // A 16x32: row lm, K = half*16 .. half*16+15
        pa.u[0] = *reinterpret_cast<const uint4*>(&Plds[wid][lm][half * 16]);
        pa.u[1] = *reinterpret_cast<const uint4*>(&Plds[wid][lm][half * 16 + 8]);

        // O += P(16x32) @ V(32x64)
        for (int t = 0; t < 4; ++t)
            o[t] = wmma_bf16(pa.v, vb[t].v, o[t]);
    }

    // normalize and store ctx in [B,H,S,Dh] bf16
    for (int r = 0; r < 8; ++r) {
        float inv = 1.0f / l_run[r];
        int s = qtile * 16 + r + half * 8;
        for (int t = 0; t < 4; ++t) {
            ctx[((size_t)bh * SEQ + s) * HDIM + t * 16 + lm] = (__bf16)(o[t][r] * inv);
        }
    }
}

// ---------------------------------------------------------------------------
// Kernel 5: out = ctx @ Wo, f32 output, LDS double-buffered.
// ctx stored as [B,H,S,Dh] (k = h*64+dh); a 32-wide k-step stays in one head.
// ---------------------------------------------------------------------------
__global__ __launch_bounds__(256)
void out_gemm_kernel(const __bf16* __restrict__ ctx, const __bf16* __restrict__ Wot,
                     float* __restrict__ out) {
    __shared__ __align__(16) __bf16 As[2][TILE_E];
    __shared__ __align__(16) __bf16 Bs[2][TILE_E];

    const int tid  = threadIdx.x;
    const int wid  = tid >> 5;
    const int lane = tid & 31;
    const int half = lane >> 4;
    const int lm   = lane & 15;
    const int wm   = wid >> 1;
    const int wn   = wid & 1;

    const int m0 = blockIdx.x * 128;
    const int n0 = blockIdx.y * 128;
    const __bf16* Bg = Wot + (size_t)n0 * DMODEL;

    auto stage = [&](int buf, int k0) {
        int h = k0 >> 6, koff = k0 & 63;      // 32-wide k-step within one head
        for (int i = 0; i < 2; ++i) {
            int t   = tid * 2 + i;
            int row = t >> 2, c = t & 3;
            int m = m0 + row, b_ = m >> 11, s = m & (SEQ - 1);
            cp16(ctx + ((size_t)(b_ * NHEAD + h) * SEQ + s) * HDIM + koff + c * 8,
                 &As[buf][row * LDSTR + c * 8]);
            cp16(Bg + (size_t)row * DMODEL + k0 + c * 8,
                 &Bs[buf][row * LDSTR + c * 8]);
        }
    };

    v8f acc[2][4];
    for (int i = 0; i < 2; ++i)
        for (int j = 0; j < 4; ++j) acc[i][j] = zero_v8f();

    stage(0, 0);
    for (int ks = 0; ks < DMODEL / KSTEP; ++ks) {
        int cur = ks & 1;
        wait_async_lds();
        __syncthreads();
        if (ks + 1 < DMODEL / KSTEP) stage(1 - cur, (ks + 1) * KSTEP);

        const __bf16* ab = &As[cur][0];
        const __bf16* bb = &Bs[cur][0];
        FragBF a[2], b[4];
        for (int i = 0; i < 2; ++i) {
            int row = ((wm * 2 + i) * 16 + lm) * LDSTR;
            a[i].u[0] = *reinterpret_cast<const uint4*>(ab + row + half * 16);
            a[i].u[1] = *reinterpret_cast<const uint4*>(ab + row + half * 16 + 8);
        }
        for (int j = 0; j < 4; ++j) {
            int row = ((wn * 4 + j) * 16 + lm) * LDSTR;
            b[j].u[0] = *reinterpret_cast<const uint4*>(bb + row + half * 16);
            b[j].u[1] = *reinterpret_cast<const uint4*>(bb + row + half * 16 + 8);
        }
        for (int i = 0; i < 2; ++i)
            for (int j = 0; j < 4; ++j)
                acc[i][j] = wmma_bf16(a[i].v, b[j].v, acc[i][j]);
        __syncthreads();
    }

    for (int i = 0; i < 2; ++i) {
        for (int j = 0; j < 4; ++j) {
            int n = n0 + (wn * 4 + j) * 16 + lm;
            for (int r = 0; r < 8; ++r) {
                int m = m0 + (wm * 2 + i) * 16 + r + half * 8;
                out[(size_t)m * DMODEL + n] = acc[i][j][r];
            }
        }
    }
}

// ---------------------------------------------------------------------------
// Host launcher
// ---------------------------------------------------------------------------
extern "C" void kernel_launch(void* const* d_in, const int* in_sizes, int n_in,
                              void* d_out, int out_size, void* d_ws, size_t ws_size,
                              hipStream_t stream) {
    const float* inputs = (const float*)d_in[0];
    const float* Wq = (const float*)d_in[1];
    const float* Wk = (const float*)d_in[2];
    const float* Wv = (const float*)d_in[3];
    const float* Wo = (const float*)d_in[4];
    float* out = (float*)d_out;

    const size_t MB = 1ull << 20;
    char* ws = (char*)d_ws;
    __bf16* Xbf = (__bf16*)(ws + 0 * MB);    // 8 MiB  [4096,1024]
    __bf16* Wqt = (__bf16*)(ws + 8 * MB);    // 2 MiB  [N,K]
    __bf16* Wkt = (__bf16*)(ws + 10 * MB);
    __bf16* Wvt = (__bf16*)(ws + 12 * MB);
    __bf16* Wot = (__bf16*)(ws + 14 * MB);
    __bf16* Qb  = (__bf16*)(ws + 16 * MB);   // 8 MiB  [B,H,S,Dh]
    __bf16* Kb  = (__bf16*)(ws + 24 * MB);   // 8 MiB  [B,H,S,Dh]
    __bf16* Vtb = (__bf16*)(ws + 32 * MB);   // 8 MiB  [B,H,Dh,S]
    __bf16* Ctx = (__bf16*)(ws + 40 * MB);   // 8 MiB  [B,H,S,Dh]

    // 1) cast activations (4 elems/thread)
    {
        int n4 = (MTOT * DMODEL) / 4;   // 1M
        cast_in_kernel<<<n4 / 256, 256, 0, stream>>>(inputs, Xbf, n4);
    }
    // 2) transpose-cast the four weight matrices
    {
        dim3 grid(DMODEL / 256, DMODEL, 4);
        wtrans_kernel<<<grid, 256, 0, stream>>>(Wq, Wk, Wv, Wo, Wqt, Wkt, Wvt, Wot);
    }
    // 3) QKV projection: 32x8 blocks of 128x128, z = which matrix
    {
        dim3 grid(MTOT / 128, DMODEL / 128, 3);
        qkv_gemm_kernel<<<grid, 256, 0, stream>>>(Xbf, Wqt, Wkt, Wvt, Qb, Kb, Vtb);
    }
    // 4) causal flash attention: 512 blocks, 8 q-tiles of one (b,h) per block
    attn_kernel<<<512, 256, 0, stream>>>(Qb, Kb, Vtb, Ctx);
    // 5) output projection -> f32 d_out
    {
        dim3 grid(MTOT / 128, DMODEL / 128);
        out_gemm_kernel<<<grid, 256, 0, stream>>>(Ctx, Wot, out);
    }
}
